// Attention_1846835937800
// MI455X (gfx1250) — matrix operand order repaired and checked by reading, not compiled, not run on hardware
//
#include <hip/hip_runtime.h>

typedef __attribute__((ext_vector_type(16))) _Float16 v16h;
typedef __attribute__((ext_vector_type(8)))  _Float16 v8h;
typedef __attribute__((ext_vector_type(4)))  _Float16 v4h;
typedef __attribute__((ext_vector_type(2)))  _Float16 v2h;
typedef __attribute__((ext_vector_type(8)))  float    v8f;
typedef __attribute__((ext_vector_type(4)))  float    v4f;
typedef __attribute__((ext_vector_type(8)))  int      v8i;

#define NHEADS 4
#define DH     64
#define EMB    256
#define NSEQ   4096
#define NBATCH 4
#define BC     32            // key tile (K dim of the PV WMMA)
#define NT     (NSEQ / BC)   // 128 key tiles

// xor-16 half-wave exchange via LDS crossbar (no LDS memory touched)
__device__ __forceinline__ int xor16i(int x, int lane) {
  return __builtin_amdgcn_ds_bpermute(((lane ^ 16) << 2), x);
}
__device__ __forceinline__ float xor16(float x, int lane) {
  return __builtin_bit_cast(
      float, xor16i(__builtin_bit_cast(int, x), lane));
}

// 16 contiguous fp32 -> v16h (with scale): B fragments (Q^T, W^T chunks)
__device__ __forceinline__ v16h cvt16(const float* p, float s) {
  v16h r;
#pragma unroll
  for (int t = 0; t < 4; ++t) {
    v4f f = *(const v4f*)(p + 4 * t);
#pragma unroll
    for (int i = 0; i < 4; ++i) r[4 * t + i] = (_Float16)(f[i] * s);
  }
  return r;
}

// 16-bit fragment from LDS: halves 0..7 <- p[0..7], halves 8..15 <- p[8..15]
__device__ __forceinline__ v16h lds_frag(const _Float16* p) {
  v8h lo = *(const v8h*)p;
  v8h hh = *(const v8h*)(p + 8);
  v16h r;
#pragma unroll
  for (int j = 0; j < 8; ++j) { r[j] = lo[j]; r[8 + j] = hh[j]; }
  return r;
}

__global__ __launch_bounds__(512) void attn_fused_kernel(
    const float* __restrict__ qg, const float* __restrict__ kg,
    const float* __restrict__ vg, const float* __restrict__ wmat,
    const float* __restrict__ bvec, float* __restrict__ out) {
  // Double-buffered staged tiles, f16:
  //   KfS[buf][h][kk][d] : 32 keys x 64 dims, row-major   (32 KB)
  //   VtS[buf][h][d][kk] : transposed, PV A-frags contiguous (32 KB)
  __shared__ alignas(16) _Float16 KfS[2][NHEADS][BC * DH];
  __shared__ alignas(16) _Float16 VtS[2][NHEADS][DH * BC];
  // 64x256 f16 attention-output tile for the fused projection:
  // aliases KfS (16384 halves each) -- only used after the final barrier.
  _Float16* Xs = &KfS[0][0][0];

  const int tid  = threadIdx.x;
  const int lane = tid & 31;
  const int wid  = tid >> 5;     // 16 waves
  const int head = wid >> 2;     // attention: wave's head
  const int qt   = wid & 3;      // attention: wave's 16-row query sub-tile
  const int hi   = lane >> 4;
  const int n16  = lane & 15;
  const int bidx = blockIdx.x >> 6;          // batch
  const int q0   = (blockIdx.x & 63) << 6;   // 64-row query block

  // staging roles: 128 threads per head
  const int hstage = tid >> 7;
  const int ts     = tid & 127;
  // K staging: (key row, 16-dim quarter)
  const int skk = ts >> 2;
  const int sq  = ts & 3;
  // V staging: 4 keys x 4 dims micro-tile, register transpose
  const int dq  = ts & 15;       // dim quad 0..15
  const int kq  = ts >> 4;       // key quad 0..7
  const float* kstage = kg + ((size_t)bidx * NSEQ) * EMB + hstage * DH;
  const float* vstage = vg + ((size_t)bidx * NSEQ) * EMB + hstage * DH;

  auto stage = [&](int buf, int k0) {
    // ---- K tile: contiguous row chunks -> f16, row-major in LDS ----
    const float* kp = kstage + (size_t)(k0 + skk) * EMB + sq * 16;
    v4f ka[4];
#pragma unroll
    for (int t = 0; t < 4; ++t) ka[t] = *(const v4f*)(kp + 4 * t);
    v8h h0, h1;
#pragma unroll
    for (int i = 0; i < 4; ++i) {
      h0[i]     = (_Float16)ka[0][i];
      h0[4 + i] = (_Float16)ka[1][i];
      h1[i]     = (_Float16)ka[2][i];
      h1[4 + i] = (_Float16)ka[3][i];
    }
    _Float16* kd = &KfS[buf][hstage][skk * DH + sq * 16];
    *(v8h*)kd = h0;
    *(v8h*)(kd + 8) = h1;
    // ---- V tile: 4x4 register transpose, packed ds_store_b64 ----
    v4f va[4];  // va[r] = V[k0 + kq*4 + r][dq*4 .. +3]
#pragma unroll
    for (int r = 0; r < 4; ++r)
      va[r] = *(const v4f*)(vstage + (size_t)(k0 + kq * 4 + r) * EMB + dq * 4);
#pragma unroll
    for (int i = 0; i < 4; ++i) {
      v4h col = {(_Float16)va[0][i], (_Float16)va[1][i],
                 (_Float16)va[2][i], (_Float16)va[3][i]};
      *(v4h*)&VtS[buf][hstage][(dq * 4 + i) * BC + kq * 4] = col;
    }
  };

  // 1/sqrt(64) * log2(e): fold softmax scale + exp2 conversion into Q
  const float SCALE = 0.125f * 1.44269504088896340736f;

  // Q fragments (B operand of S^T = K x Q^T), kept in registers
  const float* qb = qg + ((size_t)bidx * NSEQ) * EMB + head * DH;
  v16h qf[2];
#pragma unroll
  for (int dc = 0; dc < 2; ++dc)
    qf[dc] = cvt16(qb + (size_t)(q0 + qt * 16 + n16) * EMB + dc * 32 + 16 * hi,
                   SCALE);

  // O^T accumulators: C layout, N = query = n16, M = d
  v8f oacc[4];
#pragma unroll
  for (int dc = 0; dc < 4; ++dc)
#pragma unroll
    for (int j = 0; j < 8; ++j) oacc[dc][j] = 0.f;
  float mrun = -3.0e38f;
  float lrun = 0.f;

  stage(0, 0);
  __syncthreads();

  for (int kt = 0; kt < NT; ++kt) {
    const int buf = kt & 1;
    if (kt + 1 < NT) stage(buf ^ 1, (kt + 1) * BC);  // overlaps compute

    const _Float16* Kh = &KfS[buf][head][0];
    const _Float16* Vh = &VtS[buf][head][0];

    // ---- S^T = K_tile x Q^T : two 16-key groups, accumulate over d ----
    v8f st[2];
#pragma unroll
    for (int g = 0; g < 2; ++g) {
      v8f c;
#pragma unroll
      for (int j = 0; j < 8; ++j) c[j] = 0.f;
#pragma unroll
      for (int dc = 0; dc < 2; ++dc) {
        v16h a = lds_frag(Kh + (g * 16 + n16) * DH + dc * 32 + 16 * hi);
        c = __builtin_amdgcn_wmma_f32_16x16x32_f16(false, a, false, qf[dc],
                                                   (short)0, c, false, false);
      }
      st[g] = c;
    }

    // ---- online softmax (stats per-lane: query = n16) ----
    float tmax = -3.0e38f;
#pragma unroll
    for (int g = 0; g < 2; ++g)
#pragma unroll
      for (int j = 0; j < 8; ++j) tmax = fmaxf(tmax, st[g][j]);
    tmax = fmaxf(tmax, xor16(tmax, lane));
    float mnew  = fmaxf(mrun, tmax);
    // raw v_exp_f32: args <= 0, deep-negative flushes to 0 (desired)
    float alpha = __builtin_amdgcn_exp2f(mrun - mnew);
    mrun = mnew;

    float pvals[16];
    float ssum = 0.f;
#pragma unroll
    for (int g = 0; g < 2; ++g)
#pragma unroll
      for (int j = 0; j < 8; ++j) {
        float e = __builtin_amdgcn_exp2f(st[g][j] - mnew);
        pvals[g * 8 + j] = e;
        ssum += e;
      }
    ssum += xor16(ssum, lane);
    lrun = lrun * alpha + ssum;
#pragma unroll
    for (int dc = 0; dc < 4; ++dc)
#pragma unroll
      for (int j = 0; j < 8; ++j) oacc[dc][j] *= alpha;

    // ---- P^T (C layout) -> B fragment (32k x 16q) ----
    // Pack to f16 pairs, exchange only the 8 values the partner needs:
    // 4 packed ds_bpermute + dword selects.
    int own_lo[4], own_hi[4];  // keys g*16 + 8*hi + (0..7), packed pairs
#pragma unroll
    for (int i = 0; i < 4; ++i) {
      v2h a = {(_Float16)pvals[2 * i], (_Float16)pvals[2 * i + 1]};
      v2h b = {(_Float16)pvals[8 + 2 * i], (_Float16)pvals[8 + 2 * i + 1]};
      own_lo[i] = __builtin_bit_cast(int, a);
      own_hi[i] = __builtin_bit_cast(int, b);
    }
    v8i pbw;
#pragma unroll
    for (int i = 0; i < 4; ++i) {
      int pub = hi ? own_lo[i] : own_hi[i];  // what the partner needs
      int oth = xor16i(pub, lane);
      pbw[i]     = hi ? oth       : own_lo[i];  // kk = 16*hi + 0..7
      pbw[4 + i] = hi ? own_hi[i] : oth;        // kk = 16*hi + 8..15
    }
    v16h pb = __builtin_bit_cast(v16h, pbw);

    // ---- O^T += V^T_chunk x P^T (4 d-chunks, contiguous LDS A-frags) ----
#pragma unroll
    for (int dc = 0; dc < 4; ++dc) {
      v16h a = lds_frag(Vh + (dc * 16 + n16) * BC + 16 * hi);
      oacc[dc] = __builtin_amdgcn_wmma_f32_16x16x32_f16(false, a, false, pb,
                                                        (short)0, oacc[dc],
                                                        false, false);
    }

    __syncthreads();  // next stage visible; buffers safe to reuse
  }

  // ---- normalize, stage attention output to LDS (aliases KfS) ----
  float inv = 1.0f / lrun;
#pragma unroll
  for (int dc = 0; dc < 4; ++dc) {
    v8h xh;
#pragma unroll
    for (int j = 0; j < 8; ++j) xh[j] = (_Float16)(oacc[dc][j] * inv);
    // O^T chunk rows are d = dc*16 + 8*hi + j -> contiguous 8 halves
    *(v8h*)&Xs[(qt * 16 + n16) * EMB + head * DH + dc * 16 + 8 * hi] = xh;
  }
  __syncthreads();

  // ---- fused projection: Y[64 x 256] = X @ W^T + b ----
  // wave (head, qt) -> 64 cols (c0 = head*64) x 16 rows (r0 = qt*16)
  const int c0 = head * 64;
  const int r0 = qt * 16;
  v8f acc[4];
#pragma unroll
  for (int dc = 0; dc < 4; ++dc)
#pragma unroll
    for (int j = 0; j < 8; ++j) acc[dc][j] = 0.f;

#pragma unroll
  for (int ec = 0; ec < 8; ++ec) {
    v16h xa = lds_frag(&Xs[(r0 + n16) * EMB + ec * 32 + 16 * hi]);
#pragma unroll
    for (int dc = 0; dc < 4; ++dc) {
      const float* wp =
          wmat + (size_t)(c0 + dc * 16 + n16) * EMB + ec * 32 + 16 * hi;
      v16h wb = cvt16(wp, 1.0f);
      acc[dc] = __builtin_amdgcn_wmma_f32_16x16x32_f16(false, xa, false, wb,
                                                       (short)0, acc[dc],
                                                       false, false);
    }
  }

#pragma unroll
  for (int dc = 0; dc < 4; ++dc) {
    const int col = c0 + dc * 16 + n16;
    float bi = bvec[col];
#pragma unroll
    for (int j = 0; j < 8; ++j) {
      out[(size_t)(bidx * NSEQ + q0 + r0 + 8 * hi + j) * EMB + col] =
          acc[dc][j] + bi;
    }
  }
}

extern "C" void kernel_launch(void* const* d_in, const int* in_sizes, int n_in,
                              void* d_out, int out_size, void* d_ws,
                              size_t ws_size, hipStream_t stream) {
  (void)in_sizes; (void)n_in; (void)out_size; (void)d_ws; (void)ws_size;
  const float* q    = (const float*)d_in[0];
  const float* k    = (const float*)d_in[1];
  const float* v    = (const float*)d_in[2];
  const float* wout = (const float*)d_in[3];
  const float* bo   = (const float*)d_in[4];
  float* out = (float*)d_out;

  dim3 grid(NBATCH * (NSEQ / 64));  // 256 workgroups (64-row query blocks)
  dim3 block(512);                  // 16 wave32 waves: 4 heads x 4 sub-tiles
  hipLaunchKernelGGL(attn_fused_kernel, grid, block, 0, stream,
                     q, k, v, wout, bo, out);
}
